// EdgeTypeSpecGCNLayer_43215960932827
// MI455X (gfx1250) — matrix operand order repaired and checked by reading, not compiled, not run on hardware
//
#include <hip/hip_runtime.h>

#define N_NODES 50000
#define CDIM 128
#define BLK_ROWS 64   // rows of x per block in the WMMA GEMM

typedef __attribute__((ext_vector_type(8)))  float  v8f;
typedef __attribute__((ext_vector_type(8)))  __bf16 v8bf;
typedef __attribute__((ext_vector_type(16))) __bf16 v16bf;

// ---------------------------------------------------------------------------
// PCG-style hash for deterministic dropout mask (keep prob 0.8, scale 1.25)
// ---------------------------------------------------------------------------
__device__ __forceinline__ unsigned pcg_hash(unsigned v) {
    v = v * 747796405u + 2891336453u;
    unsigned w = ((v >> ((v >> 28u) + 4u)) ^ v) * 277803737u;
    return (w >> 22u) ^ w;
}

// ---------------------------------------------------------------------------
// Utility fill
// ---------------------------------------------------------------------------
__global__ void fill_kernel(float* __restrict__ p, long long n, float val) {
    long long i = (long long)blockIdx.x * blockDim.x + threadIdx.x;
    if (i < n) p[i] = val;
}

// deg[dst] += 1 per edge (deg pre-initialized to 1.0 for the self loop)
__global__ void deg_count_kernel(const long long* __restrict__ edges, long long E,
                                 float* __restrict__ deg) {
    long long e = (long long)blockIdx.x * blockDim.x + threadIdx.x;
    if (e < E) {
        long long d = edges[E + e];
        atomicAdd(&deg[d], 1.0f);
    }
}

// deg -> deg^{-1/2} in place (deg >= 1 always, self loop guarantees it)
__global__ void rsqrt_kernel(float* __restrict__ p, int n) {
    int i = blockIdx.x * blockDim.x + threadIdx.x;
    if (i < n) p[i] = rsqrtf(p[i]);
}

// ---------------------------------------------------------------------------
// xw = dropout(x) @ W  via v_wmma_f32_16x16x32_bf16
// block = 256 threads = 8 waves; block row-tile = 64 rows; wave w owns cols
// [16w,16w+16) and iterates over 4 row tiles -> 16 WMMAs per W load.
// Tail handling is hoisted to a single uniform (scalar) branch so the common
// path has unguarded, clause-able loads/stores.
// ---------------------------------------------------------------------------
__global__ __launch_bounds__(256) void gemm_dropout_wmma_kernel(
    const float* __restrict__ x, const float* __restrict__ W,
    float* __restrict__ xw, unsigned seed)
{
    __shared__ __bf16 sWt[CDIM][CDIM];       // sWt[col][k] = W[k][col], 32 KB
    __shared__ __bf16 sX[BLK_ROWS][CDIM];    // dropout-applied x tile, 16 KB

    const int  tid      = threadIdx.x;
    const int  row0     = blockIdx.x * BLK_ROWS;
    const bool full_blk = (row0 + BLK_ROWS) <= N_NODES;   // uniform per block

    // Load W (128x128 f32 -> bf16, transposed into LDS). 16384 elems / 256 thr.
    #pragma unroll
    for (int t = 0; t < (CDIM * CDIM) / 256; ++t) {
        int idx = tid + t * 256;
        int k = idx >> 7, c = idx & 127;
        sWt[c][k] = (__bf16)W[idx];
    }
    // Load x tile with fused dropout. 8192 elems / 256 thr.
    const unsigned smix = seed * 0x9E3779B9u + 0x85EBCA6Bu;
    if (full_blk) {
        #pragma unroll
        for (int t = 0; t < (BLK_ROWS * CDIM) / 256; ++t) {
            int idx = tid + t * 256;
            int r = idx >> 7, c = idx & 127;
            unsigned g = (unsigned)((row0 + r) * CDIM + c);
            float v = x[g];
            unsigned h = pcg_hash(g ^ smix);
            sX[r][c] = (__bf16)((h < 3435973836u) ? v * 1.25f : 0.0f);
        }
    } else {
        #pragma unroll
        for (int t = 0; t < (BLK_ROWS * CDIM) / 256; ++t) {
            int idx = tid + t * 256;
            int r = idx >> 7, c = idx & 127;
            int row = row0 + r;
            float v = 0.0f;
            if (row < N_NODES) {
                unsigned g = (unsigned)(row * CDIM + c);
                v = x[g];
                unsigned h = pcg_hash(g ^ smix);
                v = (h < 3435973836u) ? v * 1.25f : 0.0f;
            }
            sX[r][c] = (__bf16)v;
        }
    }
    __syncthreads();

    const int wave = tid >> 5;
    const int lane = tid & 31;
    const int half = lane >> 4;     // 0: lanes 0-15, 1: lanes 16-31
    const int mn   = lane & 15;     // row (A) / col (B) index within tile
    const int col0 = wave * 16;

    #pragma unroll
    for (int rt = 0; rt < BLK_ROWS / 16; ++rt) {   // 4 row tiles per wave
        const int m0 = rt * 16;
        v8f acc = {0.f, 0.f, 0.f, 0.f, 0.f, 0.f, 0.f, 0.f};

        #pragma unroll
        for (int kb = 0; kb < 4; ++kb) {           // K = 128 in 4 chunks of 32
            // A frag (16-bit A 16x32): lanes 0-15 K in {0..7,16..23},
            // lanes 16-31 K in {8..15,24..31}, M = lane%16
            int ka = kb * 32 + half * 8;
            v8bf alo = *(const v8bf*)&sX[m0 + mn][ka];
            v8bf ahi = *(const v8bf*)&sX[m0 + mn][ka + 16];
            v16bf a;
            #pragma unroll
            for (int j = 0; j < 8; ++j) { a[j] = alo[j]; a[j + 8] = ahi[j]; }

            // B frag (16-bit B 32x16): lane = column, lanes 0-15 K in {0..7,16..23},
            // lanes 16-31 K in {8..15,24..31} (matches A)
            int kbb = kb * 32 + half * 8;
            v8bf blo = *(const v8bf*)&sWt[col0 + mn][kbb];
            v8bf bhi = *(const v8bf*)&sWt[col0 + mn][kbb + 16];
            v16bf b;
            #pragma unroll
            for (int j = 0; j < 8; ++j) { b[j] = blo[j]; b[j + 8] = bhi[j]; }

            acc = __builtin_amdgcn_wmma_f32_16x16x32_bf16(
                /*neg_a=*/false, a, /*neg_b=*/false, b,
                /*c_mod=*/(short)0, acc, /*reuse_a=*/false, /*reuse_b=*/false);
        }

        // C/D layout: VGPR r -> (M=r, N=lane) lanes 0-15, (M=r+8, N=lane-16)
        float* base = xw + (long long)(row0 + m0 + 8 * half) * CDIM + col0 + mn;
        if (full_blk) {
            #pragma unroll
            for (int r = 0; r < 8; ++r)
                base[(long long)r * CDIM] = acc[r];
        } else {
            #pragma unroll
            for (int r = 0; r < 8; ++r) {
                if (row0 + m0 + 8 * half + r < N_NODES)
                    base[(long long)r * CDIM] = acc[r];
            }
        }
    }
}

// ---------------------------------------------------------------------------
// accum[i] = xw[i]*dinv[n]^2 + bias[c]   (self-loop message + bias, replaces
// the zero fill and saves a full xw read in the finalize pass)
// ---------------------------------------------------------------------------
__global__ void accum_init_kernel(float* __restrict__ accum,
                                  const float* __restrict__ xw,
                                  const float* __restrict__ dinv,
                                  const float* __restrict__ bias, long long NC)
{
    long long i = (long long)blockIdx.x * blockDim.x + threadIdx.x;
    if (i >= NC) return;
    int n = (int)(i >> 7);
    int c = (int)(i & 127);
    float di = dinv[n];
    accum[i] = xw[i] * di * di + bias[c];
}

// ---------------------------------------------------------------------------
// Edge scatter: one wave per edge, lane handles 4 channels (float4 gather +
// 4x global_atomic_add_f32). accum[dst] += xw[src] * dinv[src]*dinv[dst]
// ---------------------------------------------------------------------------
__global__ __launch_bounds__(256) void scatter_kernel(
    const long long* __restrict__ edges, const float* __restrict__ dinv,
    const float* __restrict__ xw, float* __restrict__ accum, long long E)
{
    long long t = (long long)blockIdx.x * blockDim.x + threadIdx.x;
    long long e = t >> 5;
    int lane = threadIdx.x & 31;
    if (e >= E) return;
    long long s = edges[e];
    long long d = edges[E + e];
    float nrm = dinv[s] * dinv[d];
    float4 v = ((const float4*)(xw + s * CDIM))[lane];
    float* ap = accum + d * CDIM + lane * 4;
    atomicAdd(ap + 0, v.x * nrm);
    atomicAdd(ap + 1, v.y * nrm);
    atomicAdd(ap + 2, v.z * nrm);
    atomicAdd(ap + 3, v.w * nrm);
}

// ---------------------------------------------------------------------------
// out += relu(accum)
// ---------------------------------------------------------------------------
__global__ void finalize_kernel(const float* __restrict__ accum,
                                float* __restrict__ out, long long NC)
{
    long long i = (long long)blockIdx.x * blockDim.x + threadIdx.x;
    if (i < NC) out[i] += fmaxf(accum[i], 0.0f);
}

// ---------------------------------------------------------------------------
// Row-wise L2 normalize: one wave per row, float4 per lane, shfl_xor reduce
// ---------------------------------------------------------------------------
__global__ __launch_bounds__(256) void l2norm_kernel(float* __restrict__ out) {
    long long t = (long long)blockIdx.x * blockDim.x + threadIdx.x;
    long long row = t >> 5;
    int lane = threadIdx.x & 31;
    if (row >= N_NODES) return;
    float4 v = ((const float4*)(out + row * CDIM))[lane];
    float ss = v.x * v.x + v.y * v.y + v.z * v.z + v.w * v.w;
    #pragma unroll
    for (int off = 16; off > 0; off >>= 1) ss += __shfl_xor(ss, off, 32);
    float inv = 1.0f / fmaxf(sqrtf(ss), 1e-12f);
    v.x *= inv; v.y *= inv; v.z *= inv; v.w *= inv;
    ((float4*)(out + row * CDIM))[lane] = v;
}

// ---------------------------------------------------------------------------
// Host driver
// ---------------------------------------------------------------------------
extern "C" void kernel_launch(void* const* d_in, const int* in_sizes, int n_in,
                              void* d_out, int out_size, void* d_ws, size_t ws_size,
                              hipStream_t stream) {
    const float*     x    = (const float*)d_in[0];
    const float*     W    = (const float*)d_in[1];
    const float*     bias = (const float*)d_in[2];
    const long long* ea   = (const long long*)d_in[3];
    const long long* eb   = (const long long*)d_in[4];
    const long long  E    = (long long)in_sizes[3] / 2;

    const long long NC = (long long)N_NODES * CDIM;

    // Workspace layout: dinv [N] | xw [N*128] | accum [N*128]
    float* dinv  = (float*)d_ws;
    float* xw    = (float*)((char*)d_ws +
                    (((size_t)N_NODES * sizeof(float) + 255) & ~(size_t)255));
    float* accum = xw + (size_t)N_NODES * CDIM;
    float* out   = (float*)d_out;

    fill_kernel<<<(unsigned)((NC + 255) / 256), 256, 0, stream>>>(out, NC, 0.0f);

    const long long* elist[2] = {ea, eb};
    for (int i = 0; i < 2; ++i) {
        // degree (with self loop) -> dinv
        fill_kernel<<<(N_NODES + 255) / 256, 256, 0, stream>>>(dinv, N_NODES, 1.0f);
        deg_count_kernel<<<(unsigned)((E + 255) / 256), 256, 0, stream>>>(elist[i], E, dinv);
        rsqrt_kernel<<<(N_NODES + 255) / 256, 256, 0, stream>>>(dinv, N_NODES);

        // dropout + x@W via WMMA (ceil(50000/64) = 782 row blocks)
        gemm_dropout_wmma_kernel<<<(N_NODES + BLK_ROWS - 1) / BLK_ROWS, 256, 0,
                                   stream>>>(x, W, xw, (unsigned)(i + 1));

        // accum = self-loop message + bias
        accum_init_kernel<<<(unsigned)((NC + 255) / 256), 256, 0, stream>>>(
            accum, xw, dinv, bias, NC);

        // edge aggregation (one wave per edge)
        scatter_kernel<<<(unsigned)((E * 32 + 255) / 256), 256, 0, stream>>>(
            elist[i], dinv, xw, accum, E);

        // relu + accumulate into out
        finalize_kernel<<<(unsigned)((NC + 255) / 256), 256, 0, stream>>>(
            accum, out, NC);
    }

    l2norm_kernel<<<(unsigned)(((long long)N_NODES * 32 + 255) / 256), 256, 0, stream>>>(out);
}
